// EmformerEncoder_11338713662092
// MI455X (gfx1250) — matrix operand order repaired and checked by reading, not compiled, not run on hardware
//
#include <hip/hip_runtime.h>

typedef __bf16 bf16;
typedef __attribute__((ext_vector_type(16))) __bf16 bf16x16;
typedef __attribute__((ext_vector_type(8)))  __bf16 bf16x8;
typedef __attribute__((ext_vector_type(8)))  float  f32x8;

#define B_  2
#define T_  2048
#define D_  512
#define H_  8
#define DH_ 64
#define F_  2048
#define M_  (B_*T_)      // 4096 rows
#define CHUNK_ 64
#define LEFT_  32
#define RIGHT_ 16
#define WMAX_  112       // chunk+left+right
#define KP_    128       // WMAX_ padded to multiple of 32
#define NC_    (T_/CHUNK_) // 32 chunks

// ---------------------------------------------------------------- helpers

__device__ __forceinline__ f32x8 zero8() {
  f32x8 v;
  #pragma unroll
  for (int i = 0; i < 8; ++i) v[i] = 0.0f;
  return v;
}

// A fragment: 16-bit A 16x32 layout — lane half selects K-offset 0/8; elements
// 0..7 at K=kk+ah, elements 8..15 at K=kk+16+ah  (interleaved per ISA table)
__device__ __forceinline__ bf16x16 frag_a(const bf16* lo, const bf16* hi) {
  bf16x8 l = *reinterpret_cast<const bf16x8*>(lo);
  bf16x8 h = *reinterpret_cast<const bf16x8*>(hi);
  return __builtin_shufflevector(l, h, 0,1,2,3,4,5,6,7,8,9,10,11,12,13,14,15);
}
// B fragment: same interleaved K layout as A; p points at kk + 8*(lane/16)
__device__ __forceinline__ bf16x16 frag_b(const bf16* p) {
  return frag_a(p, p + 16);
}

__device__ __forceinline__ f32x8 wmma_bf16(bf16x16 a, bf16x16 b, f32x8 c) {
  return __builtin_amdgcn_wmma_f32_16x16x32_bf16(false, a, false, b,
                                                 (short)0, c, false, false);
}

// ---------------------------------------------------------------- GEMM
// C[M,N] = A[M,K](bf16 rowmajor) * Bt[N,K](bf16, pre-transposed B)
// Register double-buffered: tile k+1 global loads issued while tile k's WMMAs
// run; tile k+2 prefetched into L2 via global_prefetch_b8.
// MODE 0: QKV  -> outB[b,h,t,e] = alpha*acc           (bf16)
// MODE 1: BIAS+RES -> outF[m,n] = acc + bias[n] + res[m,n]   (f32)
// MODE 2: RELU+BIAS -> outB[m,n] = max(acc+bias[n],0)        (bf16)
template<int MODE>
__global__ __launch_bounds__(256)
void gemm_bf16_wmma(const bf16* __restrict__ A, const bf16* __restrict__ Bt,
                    int M, int N, int K, float alpha,
                    const float* __restrict__ bias, const float* __restrict__ res,
                    float* __restrict__ outF, bf16* __restrict__ outB) {
  constexpr int BM = 128, BN = 128, BK = 32, PAD = 8;
  __shared__ bf16 sA[BM][BK + PAD];
  __shared__ bf16 sB[BN][BK + PAD];

  const int tid  = threadIdx.x;
  const int lane = tid & 31;
  const int wave = tid >> 5;
  const int wm   = wave >> 2;       // 0..1  (64 rows each)
  const int wn   = wave & 3;        // 0..3  (32 cols each)
  const int bM   = blockIdx.y * BM;
  const int bN   = blockIdx.x * BN;

  f32x8 acc[4][2];
  #pragma unroll
  for (int mt = 0; mt < 4; ++mt)
    #pragma unroll
    for (int nt = 0; nt < 2; ++nt) acc[mt][nt] = zero8();

  const int lrow  = tid >> 1;            // 0..127
  const int lhalf = (tid & 1) * 16;      // 0 or 16 (elements)
  const int ah    = (lane >> 4) * 8;     // A half offset
  const int bh    = (lane >> 4) * 8;     // B half offset
  const int lr    = lane & 15;

  const bf16* gA = A  + (size_t)(bM + lrow) * K + lhalf;
  const bf16* gB = Bt + (size_t)(bN + lrow) * K + lhalf;

  // prologue: first tile into registers
  float4 ra0 = reinterpret_cast<const float4*>(gA)[0];
  float4 ra1 = reinterpret_cast<const float4*>(gA)[1];
  float4 rb0 = reinterpret_cast<const float4*>(gB)[0];
  float4 rb1 = reinterpret_cast<const float4*>(gB)[1];

  for (int k0 = 0; k0 < K; k0 += BK) {
    // stage tile k0 into LDS
    *reinterpret_cast<float4*>(&sA[lrow][lhalf])     = ra0;
    *reinterpret_cast<float4*>(&sA[lrow][lhalf + 8]) = ra1;
    *reinterpret_cast<float4*>(&sB[lrow][lhalf])     = rb0;
    *reinterpret_cast<float4*>(&sB[lrow][lhalf + 8]) = rb1;
    __syncthreads();

    // issue next tile's global loads before compute (overlap with WMMA)
    if (k0 + BK < K) {
      const bf16* pA = gA + k0 + BK;
      const bf16* pB = gB + k0 + BK;
      ra0 = reinterpret_cast<const float4*>(pA)[0];
      ra1 = reinterpret_cast<const float4*>(pA)[1];
      rb0 = reinterpret_cast<const float4*>(pB)[0];
      rb1 = reinterpret_cast<const float4*>(pB)[1];
      if (k0 + 2 * BK < K) {
        __builtin_prefetch(gA + k0 + 2 * BK, 0, 1);
        __builtin_prefetch(gB + k0 + 2 * BK, 0, 1);
      }
    }

    bf16x16 af[4], bfr[2];
    #pragma unroll
    for (int mt = 0; mt < 4; ++mt) {
      const int r = wm * 64 + mt * 16 + lr;
      af[mt] = frag_a(&sA[r][ah], &sA[r][16 + ah]);
    }
    #pragma unroll
    for (int nt = 0; nt < 2; ++nt) {
      const int r = wn * 32 + nt * 16 + lr;
      bfr[nt] = frag_b(&sB[r][bh]);
    }
    #pragma unroll
    for (int mt = 0; mt < 4; ++mt)
      #pragma unroll
      for (int nt = 0; nt < 2; ++nt)
        acc[mt][nt] = wmma_bf16(af[mt], bfr[nt], acc[mt][nt]);
    __syncthreads();
  }

  // epilogue: D layout — element j -> row j + 8*(lane/16), col lane%16
  const int rj = (lane >> 4) << 3;
  #pragma unroll
  for (int mt = 0; mt < 4; ++mt) {
    #pragma unroll
    for (int nt = 0; nt < 2; ++nt) {
      #pragma unroll
      for (int j = 0; j < 8; ++j) {
        const int m = bM + wm * 64 + mt * 16 + j + rj;
        const int n = bN + wn * 32 + nt * 16 + lr;
        float v = acc[mt][nt][j] * alpha;
        if constexpr (MODE == 0) {
          const int b = m >> 11, t = m & (T_ - 1);
          const int h = n >> 6,  e = n & (DH_ - 1);
          outB[(((size_t)(b * H_ + h)) * T_ + t) * DH_ + e] = (bf16)v;
        } else if constexpr (MODE == 1) {
          v += bias[n] + res[(size_t)m * N + n];
          outF[(size_t)m * N + n] = v;
        } else {
          v += bias[n];
          v = v > 0.0f ? v : 0.0f;
          outB[(size_t)m * N + n] = (bf16)v;
        }
      }
    }
  }
}

// ---------------------------------------------------------------- LayerNorm
// one row (D_=512) per block of 256 threads; optional f32 and bf16 outputs
__global__ __launch_bounds__(256)
void layernorm_kernel(const float* __restrict__ in, const float* __restrict__ g,
                      const float* __restrict__ bta,
                      float* __restrict__ outF, bf16* __restrict__ outB) {
  __shared__ float red[256];
  const int tid = threadIdx.x;
  const size_t row = blockIdx.x;
  const float* x = in + row * D_;
  const float x0 = x[tid], x1 = x[tid + 256];

  red[tid] = x0 + x1;
  __syncthreads();
  #pragma unroll
  for (int off = 128; off > 0; off >>= 1) {
    if (tid < off) red[tid] += red[tid + off];
    __syncthreads();
  }
  const float mean = red[0] * (1.0f / D_);
  __syncthreads();
  const float d0 = x0 - mean, d1 = x1 - mean;
  red[tid] = d0 * d0 + d1 * d1;
  __syncthreads();
  #pragma unroll
  for (int off = 128; off > 0; off >>= 1) {
    if (tid < off) red[tid] += red[tid + off];
    __syncthreads();
  }
  const float r = rsqrtf(red[0] * (1.0f / D_) + 1e-3f);
  const float y0 = d0 * r * g[tid]       + bta[tid];
  const float y1 = d1 * r * g[tid + 256] + bta[tid + 256];
  if (outF) { outF[row * D_ + tid] = y0; outF[row * D_ + tid + 256] = y1; }
  if (outB) { outB[row * D_ + tid] = (bf16)y0; outB[row * D_ + tid + 256] = (bf16)y1; }
}

// ---------------------------------------------------------------- Attention
// one workgroup (128 threads, 4 waves) per (b, h, chunk). Shared key window
// per chunk: [max(0,qs-32), min(T, qs+64+16)) — whole chunk same window.
__global__ __launch_bounds__(128)
void attention_kernel(const bf16* __restrict__ q, const bf16* __restrict__ k,
                      const bf16* __restrict__ v, bf16* __restrict__ o) {
  const int tid  = threadIdx.x;
  const int lane = tid & 31;
  const int wv   = tid >> 5;
  const int c = blockIdx.x & (NC_ - 1);
  const int h = (blockIdx.x >> 5) & (H_ - 1);
  const int b = blockIdx.x >> 8;
  const int qs  = c * CHUNK_;
  const int kst = (qs - LEFT_) > 0 ? (qs - LEFT_) : 0;
  const int ken = (qs + CHUNK_ + RIGHT_) < T_ ? (qs + CHUNK_ + RIGHT_) : T_;
  const int W   = ken - kst;  // <= 112

  __shared__ bf16  sQ[64][72];
  __shared__ bf16  sK[WMAX_][72];
  __shared__ bf16  sVt[64][KP_ + 8];
  __shared__ bf16  sP[64][KP_ + 8];
  __shared__ float sS[64][120];

  const size_t bh   = (size_t)(b * H_ + h);
  const bf16* qbase = q + bh * T_ * DH_;
  const bf16* kbase = k + bh * T_ * DH_;
  const bf16* vbase = v + bh * T_ * DH_;

  { // Q chunk: 64 rows x 64 cols
    const int r = tid >> 1, half = (tid & 1) * 32;
    const float4* src = reinterpret_cast<const float4*>(
        qbase + (size_t)(qs + r) * DH_ + half);
    float4* dst = reinterpret_cast<float4*>(&sQ[r][half]);
    dst[0] = src[0]; dst[1] = src[1]; dst[2] = src[2]; dst[3] = src[3];
  }
  // K window (zero pad past W)
  for (int task = tid; task < WMAX_ * 2; task += 128) {
    const int r = task >> 1, half = (task & 1) * 32;
    float4* dst = reinterpret_cast<float4*>(&sK[r][half]);
    if (r < W) {
      const float4* src = reinterpret_cast<const float4*>(
          kbase + (size_t)(kst + r) * DH_ + half);
      dst[0] = src[0]; dst[1] = src[1]; dst[2] = src[2]; dst[3] = src[3];
    } else {
      const float4 z = make_float4(0.f, 0.f, 0.f, 0.f);
      dst[0] = z; dst[1] = z; dst[2] = z; dst[3] = z;
    }
  }
  // V^T (transposed gather, zero pad past W)
  for (int idx = tid; idx < 64 * KP_; idx += 128) {
    const int e = idx >> 7, j = idx & (KP_ - 1);
    sVt[e][j] = (j < W) ? vbase[(size_t)(kst + j) * DH_ + e] : (bf16)0.0f;
  }
  __syncthreads();

  const int ah = (lane >> 4) * 8;
  const int bhf = (lane >> 4) * 8;
  const int lr = lane & 15;
  const int rj = (lane >> 4) << 3;

  { // S = Q * K^T : wave handles 16 rows x 112 cols
    f32x8 acc[7];
    #pragma unroll
    for (int nt = 0; nt < 7; ++nt) acc[nt] = zero8();
    #pragma unroll
    for (int kk = 0; kk < DH_; kk += 32) {
      const int ar = wv * 16 + lr;
      const bf16x16 af = frag_a(&sQ[ar][kk + ah], &sQ[ar][kk + 16 + ah]);
      #pragma unroll
      for (int nt = 0; nt < 7; ++nt) {
        const bf16x16 bf_ = frag_b(&sK[nt * 16 + lr][kk + bhf]);
        acc[nt] = wmma_bf16(af, bf_, acc[nt]);
      }
    }
    #pragma unroll
    for (int nt = 0; nt < 7; ++nt)
      #pragma unroll
      for (int j = 0; j < 8; ++j) {
        const int row = wv * 16 + j + rj;
        const int col = nt * 16 + lr;
        sS[row][col] = (col < W) ? acc[nt][j] : -1e30f;
      }
  }
  __syncthreads();

  if (tid < 64) { // softmax of one row
    float mx = -1e30f;
    for (int col = 0; col < W; ++col) mx = fmaxf(mx, sS[tid][col]);
    float sum = 0.0f;
    for (int col = 0; col < W; ++col) {
      const float e = __expf(sS[tid][col] - mx);
      sS[tid][col] = e;
      sum += e;
    }
    const float inv = 1.0f / sum;
    for (int col = 0; col < KP_; ++col)
      sP[tid][col] = (col < W) ? (bf16)(sS[tid][col] * inv) : (bf16)0.0f;
  }
  __syncthreads();

  { // O = P * V : wave handles 16 rows x 64 cols
    f32x8 acc[4];
    #pragma unroll
    for (int nt = 0; nt < 4; ++nt) acc[nt] = zero8();
    #pragma unroll
    for (int kk = 0; kk < KP_; kk += 32) {
      const int ar = wv * 16 + lr;
      const bf16x16 af = frag_a(&sP[ar][kk + ah], &sP[ar][kk + 16 + ah]);
      #pragma unroll
      for (int nt = 0; nt < 4; ++nt) {
        const bf16x16 bf_ = frag_b(&sVt[nt * 16 + lr][kk + bhf]);
        acc[nt] = wmma_bf16(af, bf_, acc[nt]);
      }
    }
    #pragma unroll
    for (int nt = 0; nt < 4; ++nt)
      #pragma unroll
      for (int j = 0; j < 8; ++j) {
        const int row = wv * 16 + j + rj;
        const int col = nt * 16 + lr;
        o[((size_t)(b * T_ + qs + row)) * D_ + h * DH_ + col] = (bf16)acc[nt][j];
      }
  }
}

// ---------------------------------------------------------------- weight prep
// out[(h*64+e)*512 + d] = in[(h*512+d)*64 + e]
__global__ __launch_bounds__(256)
void convert_qkv_weight(const float* __restrict__ in, bf16* __restrict__ out) {
  const int idx = blockIdx.x * 256 + threadIdx.x;
  if (idx >= D_ * D_) return;
  const int n = idx >> 9, d = idx & 511;
  const int h = n >> 6,  e = n & 63;
  out[idx] = (bf16)in[((size_t)(h << 9) + d) * 64 + e];
}
// out[n*K + k] = in[k*N + n]
__global__ __launch_bounds__(256)
void transpose_convert(const float* __restrict__ in, bf16* __restrict__ out,
                       int N, int K) {
  const int idx = blockIdx.x * 256 + threadIdx.x;
  if (idx >= N * K) return;
  const int n = idx / K, kk = idx % K;
  out[idx] = (bf16)in[(size_t)kk * N + n];
}

// ---------------------------------------------------------------- launch
extern "C" void kernel_launch(void* const* d_in, const int* in_sizes, int n_in,
                              void* d_out, int out_size, void* d_ws, size_t ws_size,
                              hipStream_t stream) {
  (void)in_sizes; (void)n_in; (void)out_size; (void)ws_size;
  const float* x      = (const float*)d_in[0];
  const float* wq     = (const float*)d_in[2];
  const float* wk     = (const float*)d_in[3];
  const float* wv     = (const float*)d_in[4];
  const float* wp     = (const float*)d_in[5];
  const float* pb     = (const float*)d_in[6];
  const float* ln_in_g= (const float*)d_in[7];
  const float* ln_in_b= (const float*)d_in[8];
  const float* ln1_g  = (const float*)d_in[9];
  const float* ln1_b  = (const float*)d_in[10];
  const float* ln2_g  = (const float*)d_in[11];
  const float* ln2_b  = (const float*)d_in[12];
  const float* w1     = (const float*)d_in[13];
  const float* b1     = (const float*)d_in[14];
  const float* w2     = (const float*)d_in[15];
  const float* b2     = (const float*)d_in[16];

  char* ws = (char*)d_ws;
  size_t off = 0;
  auto carve = [&](size_t bytes) -> void* {
    void* p = ws + off;
    off += (bytes + 255) & ~(size_t)255;
    return p;
  };
  float* hbuf = (float*)carve((size_t)M_ * D_ * 4);
  bf16*  xn   = (bf16*) carve((size_t)M_ * D_ * 2);
  bf16*  qb   = (bf16*) carve((size_t)M_ * D_ * 2);
  bf16*  kb   = (bf16*) carve((size_t)M_ * D_ * 2);
  bf16*  vb   = (bf16*) carve((size_t)M_ * D_ * 2);
  bf16*  ob   = (bf16*) carve((size_t)M_ * D_ * 2);
  float* attn = (float*)carve((size_t)M_ * D_ * 4);
  bf16*  yb   = (bf16*) carve((size_t)M_ * D_ * 2);
  bf16*  y1   = (bf16*) carve((size_t)M_ * F_ * 2);
  float* zb   = (float*)carve((size_t)M_ * D_ * 4);
  bf16*  Wqt  = (bf16*) carve((size_t)D_ * D_ * 2);
  bf16*  Wkt  = (bf16*) carve((size_t)D_ * D_ * 2);
  bf16*  Wvt  = (bf16*) carve((size_t)D_ * D_ * 2);
  bf16*  Wpt  = (bf16*) carve((size_t)D_ * D_ * 2);
  bf16*  W1t  = (bf16*) carve((size_t)F_ * D_ * 2);
  bf16*  W2t  = (bf16*) carve((size_t)D_ * F_ * 2);

  const dim3 gGemmD(D_ / 128, M_ / 128);   // N=512
  const dim3 gGemmF(F_ / 128, M_ / 128);   // N=2048
  const float inv_sqrt_dh = 0.125f;        // 1/sqrt(64)

  for (int i = 0; i < 2; ++i) {
    const float* hcur    = (i == 0) ? x : hbuf;
    float*       ln2_out = (i == 1) ? (float*)d_out : hbuf;

    // weight conversion/transposition to bf16 Bt layout
    convert_qkv_weight<<<(D_*D_)/256, 256, 0, stream>>>(wq + (size_t)i*H_*D_*DH_, Wqt);
    convert_qkv_weight<<<(D_*D_)/256, 256, 0, stream>>>(wk + (size_t)i*H_*D_*DH_, Wkt);
    convert_qkv_weight<<<(D_*D_)/256, 256, 0, stream>>>(wv + (size_t)i*H_*D_*DH_, Wvt);
    transpose_convert<<<(D_*D_)/256, 256, 0, stream>>>(wp + (size_t)i*H_*DH_*D_, Wpt, D_, D_);
    transpose_convert<<<(F_*D_)/256, 256, 0, stream>>>(w1 + (size_t)i*D_*F_, W1t, F_, D_);
    transpose_convert<<<(D_*F_)/256, 256, 0, stream>>>(w2 + (size_t)i*F_*D_, W2t, D_, F_);

    // pre-attention LN -> bf16
    layernorm_kernel<<<M_, 256, 0, stream>>>(hcur, ln_in_g + i*D_, ln_in_b + i*D_,
                                             nullptr, xn);
    // QKV projections (q scaled by 1/sqrt(Dh))
    gemm_bf16_wmma<0><<<gGemmD, 256, 0, stream>>>(xn, Wqt, M_, D_, D_, inv_sqrt_dh,
                                                  nullptr, nullptr, nullptr, qb);
    gemm_bf16_wmma<0><<<gGemmD, 256, 0, stream>>>(xn, Wkt, M_, D_, D_, 1.0f,
                                                  nullptr, nullptr, nullptr, kb);
    gemm_bf16_wmma<0><<<gGemmD, 256, 0, stream>>>(xn, Wvt, M_, D_, D_, 1.0f,
                                                  nullptr, nullptr, nullptr, vb);
    // banded attention
    attention_kernel<<<B_ * H_ * NC_, 128, 0, stream>>>(qb, kb, vb, ob);
    // output projection + bias + residual (un-normalized input)
    gemm_bf16_wmma<1><<<gGemmD, 256, 0, stream>>>(ob, Wpt, M_, D_, D_, 1.0f,
                                                  pb + i*D_, hcur, attn, nullptr);
    // LN1 -> bf16
    layernorm_kernel<<<M_, 256, 0, stream>>>(attn, ln1_g + i*D_, ln1_b + i*D_,
                                             nullptr, yb);
    // FFN1 (ReLU) -> bf16
    gemm_bf16_wmma<2><<<gGemmF, 256, 0, stream>>>(yb, W1t, M_, F_, D_, 1.0f,
                                                  b1 + i*F_, nullptr, nullptr, y1);
    // FFN2 + bias + residual(attn_out) -> f32
    gemm_bf16_wmma<1><<<gGemmD, 256, 0, stream>>>(y1, W2t, M_, D_, F_, 1.0f,
                                                  b2 + i*D_, attn, zb, nullptr);
    // LN2 -> f32 (next layer input, or final output)
    layernorm_kernel<<<M_, 256, 0, stream>>>(zb, ln2_g + i*D_, ln2_b + i*D_,
                                             ln2_out, nullptr);
  }
}
